// AssociationLoss_5549097746526
// MI455X (gfx1250) — matrix operand order repaired and checked by reading, not compiled, not run on hardware
//
#include <hip/hip_runtime.h>
#include <hip/hip_bf16.h>
#include <math.h>

// AssociationLoss for MI455X (gfx1250, wave32, WMMA).
// logits = (prev @ cur^T) / max(|p||c|, eps); loss = mean(BCEWithLogits(logits, ids match)).
// GEMM done with error-compensated f16 split on v_wmma_f32_16x16x32_f16:
//   x = hi + lo  (hi = f16(x), lo = f16(x - hi))
//   A*B ~= Ahi*Bhi + Ahi*Blo + Alo*Bhi    (f32 accumulate)
// Epilogue (normalize + BCE + mean) fused into the tile; 67M logits never hit memory.

namespace {
constexpr int PP = 8192;   // prev rows
constexpr int CC = 8192;   // cur rows
constexpr int DD = 256;    // feature dim
constexpr float EPSV = 1e-6f;
}

typedef __attribute__((ext_vector_type(16))) _Float16 v16h;
typedef __attribute__((ext_vector_type(8)))  _Float16 v8h;
typedef __attribute__((ext_vector_type(8)))  float    v8f;

// -------- Kernel 1: row norms for both matrices + zero the accumulator --------
// One wave32 per row: 256 floats = 2 x float4 per lane, shuffle-reduce.
__global__ __launch_bounds__(256) void norms_init_kernel(
    const float* __restrict__ prev, const float* __restrict__ cur,
    float* __restrict__ norm_prev, float* __restrict__ norm_cur,
    double* __restrict__ acc) {
  if (blockIdx.x == 0 && threadIdx.x == 0) *acc = 0.0;

  const int wave = threadIdx.x >> 5;
  const int lane = threadIdx.x & 31;
  const int row  = blockIdx.x * 8 + wave;          // 0 .. 16383

  const float* src;  float* dst;  int r;
  if (row < PP) { src = prev; dst = norm_prev; r = row; }
  else          { src = cur;  dst = norm_cur;  r = row - PP; }

  const float4* f = reinterpret_cast<const float4*>(src + (size_t)r * DD);
  float4 a = f[lane];        // floats [lane*4   .. lane*4+3]
  float4 b = f[lane + 32];   // floats [128+lane*4 .. ]
  float s = a.x*a.x + a.y*a.y + a.z*a.z + a.w*a.w
          + b.x*b.x + b.y*b.y + b.z*b.z + b.w*b.w;
  #pragma unroll
  for (int off = 16; off > 0; off >>= 1) s += __shfl_down(s, off, 32);
  if (lane == 0) dst[r] = sqrtf(s);
}

// -------- Kernel 2: WMMA GEMM + fused cosine/BCE epilogue + reduction --------
// Block = 8 wave32s. Block tile: 128 (M) x 64 (N). Wave w: rows [wave*16,+16), all 64 cols.
// K loop: 8 chunks of 32. B chunk (64x32 f32) staged in LDS as hi/lo f16,
// row stride padded to 40 halfs (80 B: 16B aligned, bank-conflict-free for 16-row reads).
__global__ __launch_bounds__(256) void assoc_gemm_kernel(
    const float* __restrict__ prev, const float* __restrict__ cur,
    const long long* __restrict__ pid, const long long* __restrict__ cid,
    const float* __restrict__ norm_prev, const float* __restrict__ norm_cur,
    double* __restrict__ acc) {
  __shared__ _Float16 s_bhi[64][40];
  __shared__ _Float16 s_blo[64][40];
  __shared__ float    s_part[8];

  const int tid  = threadIdx.x;
  const int wave = tid >> 5;
  const int lane = tid & 31;
  const int half = lane >> 4;     // 0/1: which K-half of the fragment this lane holds
  const int l16  = lane & 15;

  const int mBase = blockIdx.y * 128 + wave * 16;  // this wave's 16 prev-rows
  const int nBase = blockIdx.x * 64;               // this block's 64 cur-rows

  v8f accv[4] = {};   // four 16x16 f32 accumulators (N-tiles 0..3)

  for (int kc = 0; kc < DD; kc += 32) {
    // ---- cooperative B staging: 64 rows x 32 K, split into hi/lo f16 ----
    {
      const int n  = tid >> 2;            // 0..63
      const int kk = (tid & 3) << 3;      // 0,8,16,24
      const float4* g = reinterpret_cast<const float4*>(
          cur + (size_t)(nBase + n) * DD + kc + kk);
      float4 v0 = g[0];
      float4 v1 = g[1];
      float vf[8] = {v0.x, v0.y, v0.z, v0.w, v1.x, v1.y, v1.z, v1.w};
      #pragma unroll
      for (int i = 0; i < 8; ++i) {
        float fv = vf[i];
        _Float16 h = (_Float16)fv;
        s_bhi[n][kk + i] = h;
        s_blo[n][kk + i] = (_Float16)(fv - (float)h);
      }
    }
    __syncthreads();

    // ---- A fragment: 16 K-floats per lane per the 16-bit A 16x32 layout ----
    // lane (half,l16): K = {8*half+0..7} U {16+8*half+0..7}, M = l16.
    const float* aRow = prev + (size_t)(mBase + l16) * DD + kc;
    if (kc + 32 < DD) __builtin_prefetch(aRow + 32, 0, 1);  // next K-chunk (L2 resident)
    const float4* ap0 = reinterpret_cast<const float4*>(aRow + 8 * half);
    const float4* ap1 = reinterpret_cast<const float4*>(aRow + 16 + 8 * half);
    float4 a0 = ap0[0], a1 = ap0[1];
    float4 a2 = ap1[0], a3 = ap1[1];
    float af[16] = {a0.x, a0.y, a0.z, a0.w, a1.x, a1.y, a1.z, a1.w,
                    a2.x, a2.y, a2.z, a2.w, a3.x, a3.y, a3.z, a3.w};
    v16h ahi, alo;
    #pragma unroll
    for (int i = 0; i < 16; ++i) {
      _Float16 h = (_Float16)af[i];
      ahi[i] = h;
      alo[i] = (_Float16)(af[i] - (float)h);
    }

    // ---- 4 N-tiles, 3 WMMAs each (hi*hi + hi*lo + lo*hi) ----
    #pragma unroll
    for (int nt = 0; nt < 4; ++nt) {
      const int nLocal = nt * 16 + l16;      // B lane holds N = l16, K-half by `half`
      const v8h* bph = reinterpret_cast<const v8h*>(&s_bhi[nLocal][8 * half]);
      const v8h* bpl = reinterpret_cast<const v8h*>(&s_blo[nLocal][8 * half]);
      v8h bh0 = bph[0], bh1 = bph[2];
      v8h bl0 = bpl[0], bl1 = bpl[2];
      v16h bhi, blo;
      #pragma unroll
      for (int i = 0; i < 8; ++i) {
        bhi[i] = bh0[i]; bhi[8 + i] = bh1[i];
        blo[i] = bl0[i]; blo[8 + i] = bl1[i];
      }
      accv[nt] = __builtin_amdgcn_wmma_f32_16x16x32_f16(
          false, ahi, false, bhi, (short)0, accv[nt], false, false);
      accv[nt] = __builtin_amdgcn_wmma_f32_16x16x32_f16(
          false, ahi, false, blo, (short)0, accv[nt], false, false);
      accv[nt] = __builtin_amdgcn_wmma_f32_16x16x32_f16(
          false, alo, false, bhi, (short)0, accv[nt], false, false);
    }
    __syncthreads();   // protect LDS before next staging overwrite
  }

  // ---- fused epilogue: cosine normalize + BCE-with-logits, per C/D layout ----
  // accv[nt][r] : M = mBase + r + 8*half, N = nBase + nt*16 + l16
  float ncs[4];
  long long cids[4];
  #pragma unroll
  for (int nt = 0; nt < 4; ++nt) {
    const int col = nBase + nt * 16 + l16;
    ncs[nt]  = norm_cur[col];
    cids[nt] = cid[col];
  }

  float lane_sum = 0.0f;
  #pragma unroll
  for (int r = 0; r < 8; ++r) {
    const int row = mBase + r + 8 * half;
    const float np = norm_prev[row];
    const long long pv = pid[row];
    #pragma unroll
    for (int nt = 0; nt < 4; ++nt) {
      const float denom = fmaxf(np * ncs[nt], EPSV);
      const float x = accv[nt][r] / denom;
      const float y = (pv == cids[nt]) ? 1.0f : 0.0f;
      lane_sum += fmaxf(x, 0.0f) - x * y + log1pf(expf(-fabsf(x)));
    }
  }

  // wave32 shuffle reduce -> LDS -> one f64 atomic per block
  #pragma unroll
  for (int off = 16; off > 0; off >>= 1) lane_sum += __shfl_down(lane_sum, off, 32);
  if (lane == 0) s_part[wave] = lane_sum;
  __syncthreads();
  if (tid == 0) {
    float t = 0.0f;
    #pragma unroll
    for (int i = 0; i < 8; ++i) t += s_part[i];
    atomicAdd(acc, (double)t);   // global_atomic_add_f64
  }
}

// -------- Kernel 3: mean --------
__global__ void finalize_kernel(const double* __restrict__ acc, float* __restrict__ out) {
  out[0] = (float)(acc[0] / ((double)PP * (double)CC));
}

extern "C" void kernel_launch(void* const* d_in, const int* in_sizes, int n_in,
                              void* d_out, int out_size, void* d_ws, size_t ws_size,
                              hipStream_t stream) {
  const float*     prev = (const float*)d_in[0];        // [P, D] f32
  const float*     cur  = (const float*)d_in[1];        // [C, D] f32
  const long long* pid  = (const long long*)d_in[2];    // [P] int64
  const long long* cid  = (const long long*)d_in[3];    // [C] int64
  float* out = (float*)d_out;                           // scalar f32

  // ws layout: [0..8) f64 accumulator | [16..) norm_prev f32[P] | norm_cur f32[C]
  char*   ws        = (char*)d_ws;
  double* acc       = (double*)ws;
  float*  norm_prev = (float*)(ws + 16);
  float*  norm_cur  = norm_prev + PP;

  // 16384 rows total, 8 waves (rows) per 256-thread block
  norms_init_kernel<<<dim3((PP + CC) / 8), dim3(256), 0, stream>>>(
      prev, cur, norm_prev, norm_cur, acc);

  // grid: x = C/64 N-blocks, y = P/128 M-blocks
  assoc_gemm_kernel<<<dim3(CC / 64, PP / 128), dim3(256), 0, stream>>>(
      prev, cur, pid, cid, norm_prev, norm_cur, acc);

  finalize_kernel<<<1, 1, 0, stream>>>(acc, out);
}
